// PFEB_10591389352540
// MI455X (gfx1250) — matrix operand order repaired and checked by reading, not compiled, not run on hardware
//
#include <hip/hip_runtime.h>
#include <hip/hip_bf16.h>

typedef __attribute__((ext_vector_type(16))) _Float16 v16h;
typedef __attribute__((ext_vector_type(8)))  _Float16 v8h;
typedef __attribute__((ext_vector_type(8)))  float    v8f;
typedef __attribute__((ext_vector_type(4)))  int      v4i;

#define NPIX 9216   // 96*96
#define NP12 144    // 12*12
#define CCH  256
#define C8   32

// ---------------- K1: multi-scale fusion at 12x12 ----------------
// f[p,c] = ( relu(bn(conv4(x2))) + relu(bn(conv2(x3))) + x4 ) * dw1_w + dw1_b
__global__ void k_fuse12(const float* __restrict__ x2, const float* __restrict__ x3,
                         const float* __restrict__ x4,
                         const float* __restrict__ d4_w, const float* __restrict__ d4_b,
                         const float* __restrict__ d4_g, const float* __restrict__ d4_be,
                         const float* __restrict__ d4_m, const float* __restrict__ d4_v,
                         const float* __restrict__ d2_w, const float* __restrict__ d2_b,
                         const float* __restrict__ d2_g, const float* __restrict__ d2_be,
                         const float* __restrict__ d2_m, const float* __restrict__ d2_v,
                         const float* __restrict__ dw1_w, const float* __restrict__ dw1_b,
                         float* __restrict__ f_ws)
{
    int p = blockIdx.x;          // 0..143
    int c = threadIdx.x;         // 0..255
    int py = p / 12, px = p % 12;

    float s4 = 0.f;
    const float* w4 = d4_w + c * 4096;
    for (int ci = 0; ci < 256; ++ci) {
        const float* xp = x2 + ci * 2304 + (py * 4) * 48 + px * 4;
        const float* wp = w4 + ci * 16;
        #pragma unroll
        for (int ky = 0; ky < 4; ++ky)
            #pragma unroll
            for (int kx = 0; kx < 4; ++kx)
                s4 = fmaf(xp[ky * 48 + kx], wp[ky * 4 + kx], s4);
    }
    s4 += d4_b[c];
    float a = (s4 - d4_m[c]) * rsqrtf(d4_v[c] + 1e-5f) * d4_g[c] + d4_be[c];
    a = fmaxf(a, 0.f);

    float s2 = 0.f;
    const float* w2 = d2_w + c * 1024;
    for (int ci = 0; ci < 256; ++ci) {
        const float* xp = x3 + ci * 576 + (py * 2) * 24 + px * 2;
        const float* wp = w2 + ci * 4;
        s2 = fmaf(xp[0],  wp[0], s2);
        s2 = fmaf(xp[1],  wp[1], s2);
        s2 = fmaf(xp[24], wp[2], s2);
        s2 = fmaf(xp[25], wp[3], s2);
    }
    s2 += d2_b[c];
    float b = (s2 - d2_m[c]) * rsqrtf(d2_v[c] + 1e-5f) * d2_g[c] + d2_be[c];
    b = fmaxf(b, 0.f);

    float f = (a + b + x4[c * NP12 + p]) * dw1_w[c] + dw1_b[c];
    f_ws[p * 256 + c] = f;
}

// ---------------- K2: up = relu(bn(up_w@f)), qs = q_w@up + q_b  (per 12x12 pixel) ----
__global__ void k_up_qs(const float* __restrict__ f_ws,
                        const float* __restrict__ up_w, const float* __restrict__ up_b,
                        const float* __restrict__ up_g, const float* __restrict__ up_be,
                        const float* __restrict__ up_m, const float* __restrict__ up_v,
                        const float* __restrict__ q_w, const float* __restrict__ q_b,
                        float* __restrict__ qs_ws)
{
    __shared__ float fs[256];
    __shared__ float us[256];
    int p = blockIdx.x;
    int c = threadIdx.x;
    fs[c] = f_ws[p * 256 + c];
    __syncthreads();
    float s = 0.f;
    const float* w = up_w + c * 256;
    for (int i = 0; i < 256; ++i) s = fmaf(w[i], fs[i], s);
    s += up_b[c];
    s = (s - up_m[c]) * rsqrtf(up_v[c] + 1e-5f) * up_g[c] + up_be[c];
    us[c] = fmaxf(s, 0.f);
    __syncthreads();
    if (c < 32) {
        float q = 0.f;
        const float* qw = q_w + c * 256;
        for (int i = 0; i < 256; ++i) q = fmaf(qw[i], us[i], q);
        qs_ws[c * NP12 + p] = q + q_b[c];   // (32, 12, 12)
    }
}

// ---------------- K3: bilinear x8 upsample of qs (32ch), write q_t[n*32+d] f16 -------
__global__ void k_upsample_q(const float* __restrict__ qs_ws, _Float16* __restrict__ q_t)
{
    int idx = blockIdx.x * blockDim.x + threadIdx.x;  // 294912
    int d = idx & 31;
    int n = idx >> 5;
    int y = n / 96, x = n % 96;
    float sy = (y + 0.5f) * 0.125f - 0.5f;
    float sx = (x + 0.5f) * 0.125f - 0.5f;
    int y0 = (int)floorf(sy); float fy = sy - (float)y0;
    int x0 = (int)floorf(sx); float fx = sx - (float)x0;
    int y0c = max(y0, 0), y1c = min(y0 + 1, 11);
    int x0c = max(x0, 0), x1c = min(x0 + 1, 11);
    const float* qd = qs_ws + d * NP12;
    float v00 = qd[y0c * 12 + x0c], v01 = qd[y0c * 12 + x1c];
    float v10 = qd[y1c * 12 + x0c], v11 = qd[y1c * 12 + x1c];
    float val = (1.f - fy) * ((1.f - fx) * v00 + fx * v01)
              +        fy  * ((1.f - fx) * v10 + fx * v11);
    q_t[idx] = (_Float16)val;   // idx == n*32 + d
}

// ---------------- K4: k/v 1x1 convs of x1; k_t[n*32+d], v_tmp[(d,s)] flat ------------
__global__ void k_kv(const float* __restrict__ x1,
                     const float* __restrict__ k_w, const float* __restrict__ k_b,
                     const float* __restrict__ v_w, const float* __restrict__ v_b,
                     _Float16* __restrict__ k_t, _Float16* __restrict__ v_tmp)
{
    int idx = blockIdx.x * blockDim.x + threadIdx.x;  // 294912
    int s = idx % NPIX;
    int d = idx / NPIX;
    float kk = 0.f, vv = 0.f;
    const float* kw = k_w + d * 256;
    const float* vw = v_w + d * 256;
    for (int c = 0; c < 256; ++c) {
        float xv = x1[c * NPIX + s];
        kk = fmaf(kw[c], xv, kk);
        vv = fmaf(vw[c], xv, vv);
    }
    k_t[s * 32 + d]    = (_Float16)(kk + k_b[d]);
    v_tmp[d * NPIX + s] = (_Float16)(vv + v_b[d]);
}

// ---------------- K5: vt[c*N+n] = vf[n][c] = v_tmp_flat[n*32+c]  (faithful alias) ----
__global__ void k_vt(const _Float16* __restrict__ v_tmp, _Float16* __restrict__ vt)
{
    int i = blockIdx.x * blockDim.x + threadIdx.x;    // 294912
    int c = i / NPIX, n = i % NPIX;
    vt[i] = v_tmp[n * 32 + c];
    (void)c;
}

// ---------------- K6: flash attention with WMMA ------------------
// A 16x32 f16 operand: lane L -> row = L%16; halves = K{0..15} (L<16) or {16..31}
__device__ __forceinline__ v16h load32(const _Float16* p, int o8)
{
    v8h lo = *(const v8h*)(p + 2 * o8);
    v8h hi = *(const v8h*)(p + 2 * o8 + 8);
    return __builtin_shufflevector(lo, hi, 0,1,2,3,4,5,6,7,8,9,10,11,12,13,14,15);
}
__device__ __forceinline__ float swap16f(float x)
{
    return __builtin_bit_cast(float,
        __builtin_amdgcn_ds_swizzle(__builtin_bit_cast(int, x), 0x401f)); // XOR 0x10
}

__global__ void __launch_bounds__(128) k_attn(const _Float16* __restrict__ q_t,
                                              const _Float16* __restrict__ k_t,
                                              const _Float16* __restrict__ vt,
                                              float* __restrict__ ao)
{
    const int lane  = threadIdx.x & 31;
    const int wv    = threadIdx.x >> 5;
    const int qbase = (blockIdx.x * 4 + wv) * 16;
    const int m     = lane & 15;
    const int o8    = (lane < 16) ? 0 : 8;

    // B operand Q^T (32x16): lane -> col m; halves = d 0..15 (lane<16) or 16..31
    v16h bq = *(const v16h*)(q_t + (qbase + m) * 32 + ((lane < 16) ? 0 : 16));

    v8f acc0 = {}; v8f acc1 = {};
    float run_max = -1e30f;
    float run_sum = 0.f;

    for (int kb = 0; kb < NPIX; kb += 32) {
        __builtin_prefetch(k_t + (kb + 32 + m) * 32, 0, 1);
        __builtin_prefetch(vt + m * NPIX + kb + 32, 0, 1);

        // S^T tiles = K_block(16x32) x Q^T(32x16)
        v16h ak0 = load32(k_t + (kb +      m) * 32, o8);
        v16h ak1 = load32(k_t + (kb + 16 + m) * 32, o8);
        v8f cz = {};
        v8f s0 = __builtin_amdgcn_wmma_f32_16x16x32_f16(false, ak0, false, bq, (short)0, cz, false, false);
        v8f s1 = __builtin_amdgcn_wmma_f32_16x16x32_f16(false, ak1, false, bq, (short)0, cz, false, false);

        // block max over this lane's 16 keys, then partner (lane^16) combine
        float mx = s0[0];
        #pragma unroll
        for (int i = 1; i < 8; ++i) mx = fmaxf(mx, s0[i]);
        #pragma unroll
        for (int i = 0; i < 8; ++i) mx = fmaxf(mx, s1[i]);
        mx = fmaxf(mx, swap16f(mx));
        float nmax  = fmaxf(run_max, mx);
        float scale = __expf(run_max - nmax);

        float e0[8], e1[8], ssum = 0.f;
        #pragma unroll
        for (int i = 0; i < 8; ++i) { e0[i] = __expf(s0[i] - nmax); ssum += e0[i]; }
        #pragma unroll
        for (int i = 0; i < 8; ++i) { e1[i] = __expf(s1[i] - nmax); ssum += e1[i]; }
        ssum += swap16f(ssum);
        run_sum = run_sum * scale + ssum;
        run_max = nmax;
        #pragma unroll
        for (int i = 0; i < 8; ++i) { acc0[i] *= scale; acc1[i] *= scale; }

        // Build P^T B-operand (32x16): lane -> col m, halves = keys 0..15 / 16..31.
        // Lane<16 owns keys {0..7,16..23}; partner lane^16 owns {8..15,24..31}.
        v8h h0, h1;
        #pragma unroll
        for (int i = 0; i < 8; ++i) { h0[i] = (_Float16)e0[i]; h1[i] = (_Float16)e1[i]; }
        v4i h0i = __builtin_bit_cast(v4i, h0);
        v4i h1i = __builtin_bit_cast(v4i, h1);
        v4i h0p, h1p;
        #pragma unroll
        for (int i = 0; i < 4; ++i) {
            h0p[i] = __builtin_amdgcn_ds_swizzle(h0i[i], 0x401f);
            h1p[i] = __builtin_amdgcn_ds_swizzle(h1i[i], 0x401f);
        }
        v8h lo, hi;
        if (lane < 16) { lo = h0;                            hi = __builtin_bit_cast(v8h, h0p); }
        else           { lo = __builtin_bit_cast(v8h, h1p);  hi = h1; }
        v16h bp = __builtin_shufflevector(lo, hi, 0,1,2,3,4,5,6,7,8,9,10,11,12,13,14,15);

        // ao^T tiles = V^T(16x32) x P^T(32x16); vt rows contiguous in k
        v16h av0 = load32(vt + m * NPIX + kb,        o8);
        v16h av1 = load32(vt + (m + 16) * NPIX + kb, o8);
        acc0 = __builtin_amdgcn_wmma_f32_16x16x32_f16(false, av0, false, bp, (short)0, acc0, false, false);
        acc1 = __builtin_amdgcn_wmma_f32_16x16x32_f16(false, av1, false, bp, (short)0, acc1, false, false);
    }

    float inv = 1.0f / run_sum;
    int row = (qbase + m) * 32;
    #pragma unroll
    for (int i = 0; i < 8; ++i) {
        int c = i + ((lane < 16) ? 0 : 8);
        ao[row + c]      = acc0[i] * inv;
        ao[row + 16 + c] = acc1[i] * inv;
    }
}

// ---------------- K7: out = last_w @ ao^T + last_b + x1 ----------------
__global__ void k_out(const float* __restrict__ ao,
                      const float* __restrict__ last_w, const float* __restrict__ last_b,
                      const float* __restrict__ x1, float* __restrict__ out)
{
    int idx = blockIdx.x * blockDim.x + threadIdx.x;  // 2359296
    int s = idx % NPIX;
    int o = idx / NPIX;
    const float* w = last_w + o * 32;
    const float* a = ao + s * 32;
    float r = last_b[o];
    #pragma unroll
    for (int c = 0; c < 32; ++c) r = fmaf(w[c], a[c], r);
    out[idx] = r + x1[idx];
}

extern "C" void kernel_launch(void* const* d_in, const int* in_sizes, int n_in,
                              void* d_out, int out_size, void* d_ws, size_t ws_size,
                              hipStream_t stream)
{
    const float* x1    = (const float*)d_in[0];
    const float* x2    = (const float*)d_in[1];
    const float* x3    = (const float*)d_in[2];
    const float* x4    = (const float*)d_in[3];
    const float* d4_w  = (const float*)d_in[4];
    const float* d4_b  = (const float*)d_in[5];
    const float* d4_g  = (const float*)d_in[6];
    const float* d4_be = (const float*)d_in[7];
    const float* d4_m  = (const float*)d_in[8];
    const float* d4_v  = (const float*)d_in[9];
    const float* d2_w  = (const float*)d_in[10];
    const float* d2_b  = (const float*)d_in[11];
    const float* d2_g  = (const float*)d_in[12];
    const float* d2_be = (const float*)d_in[13];
    const float* d2_m  = (const float*)d_in[14];
    const float* d2_v  = (const float*)d_in[15];
    const float* dw1_w = (const float*)d_in[16];
    const float* dw1_b = (const float*)d_in[17];
    const float* up_w  = (const float*)d_in[18];
    const float* up_b  = (const float*)d_in[19];
    const float* up_g  = (const float*)d_in[20];
    const float* up_be = (const float*)d_in[21];
    const float* up_m  = (const float*)d_in[22];
    const float* up_v  = (const float*)d_in[23];
    const float* q_w   = (const float*)d_in[24];
    const float* q_b   = (const float*)d_in[25];
    const float* k_w   = (const float*)d_in[26];
    const float* k_b   = (const float*)d_in[27];
    const float* v_w   = (const float*)d_in[28];
    const float* v_b   = (const float*)d_in[29];
    const float* last_w = (const float*)d_in[30];
    const float* last_b = (const float*)d_in[31];
    float* out = (float*)d_out;

    // workspace layout (bytes)
    char* ws = (char*)d_ws;
    float*    f_ws  = (float*)   (ws + 0);         // 144*256 f32  = 147456
    float*    qs_ws = (float*)   (ws + 147456);    // 32*144 f32   = 18432
    _Float16* q_t   = (_Float16*)(ws + 165888);    // 294912 f16   = 589824
    _Float16* k_t   = (_Float16*)(ws + 755712);    // 294912 f16
    _Float16* v_tmp = (_Float16*)(ws + 1345536);   // 294912 f16
    _Float16* vt    = (_Float16*)(ws + 1935360);   // 294912 f16
    float*    ao    = (float*)   (ws + 2525184);   // 294912 f32   = 1179648  (total 3704832 B)

    k_fuse12<<<NP12, 256, 0, stream>>>(x2, x3, x4,
        d4_w, d4_b, d4_g, d4_be, d4_m, d4_v,
        d2_w, d2_b, d2_g, d2_be, d2_m, d2_v,
        dw1_w, dw1_b, f_ws);
    k_up_qs<<<NP12, 256, 0, stream>>>(f_ws, up_w, up_b, up_g, up_be, up_m, up_v,
                                      q_w, q_b, qs_ws);
    k_upsample_q<<<1152, 256, 0, stream>>>(qs_ws, q_t);
    k_kv<<<1152, 256, 0, stream>>>(x1, k_w, k_b, v_w, v_b, k_t, v_tmp);
    k_vt<<<1152, 256, 0, stream>>>(v_tmp, vt);
    k_attn<<<NP12, 128, 0, stream>>>(q_t, k_t, vt, ao);
    k_out<<<9216, 256, 0, stream>>>(ao, last_w, last_b, x1, out);
}
